// BacktrackingNetwork_29403346109073
// MI455X (gfx1250) — matrix operand order repaired and checked by reading, not compiled, not run on hardware
//
#include <hip/hip_runtime.h>

#define NB 2
#define NN 50000
#define NE 250000
#define ND 64
#define NTT 16
#define NL 3

typedef _Float16 v8h  __attribute__((ext_vector_type(8)));
typedef _Float16 v16h __attribute__((ext_vector_type(16)));
typedef float    v8f  __attribute__((ext_vector_type(8)));

__device__ __forceinline__ v16h load_frag16(const _Float16* p0, const _Float16* p1) {
    v16h r;
    ((v8h*)&r)[0] = *(const v8h*)p0;
    ((v8h*)&r)[1] = *(const v8h*)p1;
    return r;
}

// ---- weight prep: f32 (k-major) -> f16 transposed (n-major) -----------------
__global__ void wconvert_kernel(const float* __restrict__ feW, const float* __restrict__ fvW,
                                _Float16* __restrict__ feWt, _Float16* __restrict__ fvWt) {
    int idx = blockIdx.x * blockDim.x + threadIdx.x;
    const int feTotal = NL * ND * 2 * ND;      // l, n, k(0..127)
    if (idx < feTotal) {
        int l = idx / (ND * 2 * ND);
        int r = idx % (ND * 2 * ND);
        int n = r / (2 * ND);
        int k = r % (2 * ND);
        feWt[idx] = (_Float16)feW[((size_t)l * 2 * ND + k) * ND + n];
    }
    const int fvTotal = NL * ND * ND;          // l, n, k(0..63)
    if (idx < fvTotal) {
        int l = idx / (ND * ND);
        int r = idx % (ND * ND);
        int n = r / ND;
        int k = r % ND;
        fvWt[idx] = (_Float16)fvW[((size_t)l * ND + k) * ND + n];
    }
}

// ---- h0 = relu(x @ pv_W + pv_b), stored f16 ---------------------------------
__global__ void init_h_kernel(const float* __restrict__ x, const float* __restrict__ pvW,
                              const float* __restrict__ pvb, _Float16* __restrict__ h) {
    long long idx = (long long)blockIdx.x * blockDim.x + threadIdx.x;   // over B*N*D
    if (idx >= (long long)NB * NN * ND) return;
    int d = (int)(idx & (ND - 1));
    long long node = idx >> 6;
    const float* xp = x + node * 3;
    float v = xp[0] * pvW[d] + xp[1] * pvW[ND + d] + xp[2] * pvW[2 * ND + d] + pvb[d];
    h[idx] = (_Float16)fmaxf(v, 0.f);
}

// ---- g0 = relu(edge_attr @ pe_W + pe_b), broadcast to both batches, f16 -----
__global__ void init_g_kernel(const float* __restrict__ ea, const float* __restrict__ peW,
                              const float* __restrict__ peb, _Float16* __restrict__ g) {
    long long idx = (long long)blockIdx.x * blockDim.x + threadIdx.x;   // over E*D
    if (idx >= (long long)NE * ND) return;
    int d = (int)(idx & (ND - 1));
    long long e = idx >> 6;
    float v = peb[d];
    #pragma unroll
    for (int t = 0; t < NTT; ++t) v += ea[e * NTT + t] * peW[t * ND + d];
    _Float16 hv = (_Float16)fmaxf(v, 0.f);
    g[idx] = hv;
    g[idx + (long long)NE * ND] = hv;
}

// ---- edge MLP: g = relu([g, h[src]] @ We + be); scatter-add into agg[dst] ---
// one wave per 16-edge tile; 4 K-steps x 4 N-tiles of v_wmma_f32_16x16x32_f16
__global__ void edge_mlp_kernel(_Float16* __restrict__ g,
                                const _Float16* __restrict__ h,
                                const int* __restrict__ src,
                                const int* __restrict__ dst,
                                const _Float16* __restrict__ Wt,   // (64 n) x (128 k), n-major
                                const float* __restrict__ bias,    // (64)
                                float* __restrict__ agg) {
    const int tilesPerBatch = NE / 16;
    int wave = (blockIdx.x * blockDim.x + threadIdx.x) >> 5;
    if (wave >= NB * tilesPerBatch) return;            // wave-uniform: EXEC stays all-ones
    int lane = threadIdx.x & 31;
    int b = wave / tilesPerBatch;
    int ebase = (wave % tilesPerBatch) * 16;
    int row = lane & 15;
    int hi  = lane >> 4;
    int e = ebase + row;
    int srcid = src[e];
    const _Float16* grow = g + ((size_t)b * NE + e) * ND;
    const _Float16* hrow = h + ((size_t)b * NN + srcid) * ND;

    v8f acc[4] = {};
    #pragma unroll
    for (int ks = 0; ks < 4; ++ks) {
        int kb = ks * 32 + hi * 8;                     // A: runs at kb, kb+16 (8 halves each)
        const _Float16* ap = (ks < 2) ? (grow + kb) : (hrow + (kb - 64));
        v16h A = load_frag16(ap, ap + 16);
        int kwb = ks * 32 + hi * 8;                    // B: runs at kwb, kwb+16 (8 halves each)
        #pragma unroll
        for (int nt = 0; nt < 4; ++nt) {
            const _Float16* bp = Wt + (size_t)(nt * 16 + row) * (2 * ND) + kwb;
            v16h Bm = load_frag16(bp, bp + 16);
            acc[nt] = __builtin_amdgcn_wmma_f32_16x16x32_f16(
                false, A, false, Bm, (short)0, acc[nt], false, false);
        }
    }
    int dstid[8];
    #pragma unroll
    for (int r = 0; r < 8; ++r) dstid[r] = dst[ebase + hi * 8 + r];
    #pragma unroll
    for (int nt = 0; nt < 4; ++nt) {
        int col = nt * 16 + row;
        float bc = bias[col];
        #pragma unroll
        for (int r = 0; r < 8; ++r) {
            float v = acc[nt][r] + bc;
            v = v > 0.f ? v : 0.f;
            int erow = ebase + hi * 8 + r;
            g[((size_t)b * NE + erow) * ND + col] = (_Float16)v;
            unsafeAtomicAdd(&agg[((size_t)b * NN + dstid[r]) * ND + col], v);
        }
    }
}

// ---- node MLP: h = relu(relu(h @ Wv + bv) + agg), in place (f16) ------------
__global__ void node_mlp_kernel(_Float16* __restrict__ h,
                                const _Float16* __restrict__ Wt,   // (64 n) x (64 k), n-major
                                const float* __restrict__ bias,
                                const float* __restrict__ agg) {
    const int tilesPerBatch = NN / 16;
    int wave = (blockIdx.x * blockDim.x + threadIdx.x) >> 5;
    if (wave >= NB * tilesPerBatch) return;
    int lane = threadIdx.x & 31;
    int b = wave / tilesPerBatch;
    int nbase = (wave % tilesPerBatch) * 16;
    int row = lane & 15;
    int hi  = lane >> 4;
    const _Float16* hrow = h + ((size_t)b * NN + nbase + row) * ND;

    v8f acc[4] = {};
    #pragma unroll
    for (int ks = 0; ks < 2; ++ks) {
        int kb = ks * 32 + hi * 8;
        v16h A = load_frag16(hrow + kb, hrow + kb + 16);
        int kwb = ks * 32 + hi * 8;
        #pragma unroll
        for (int nt = 0; nt < 4; ++nt) {
            const _Float16* bp = Wt + (size_t)(nt * 16 + row) * ND + kwb;
            v16h Bm = load_frag16(bp, bp + 16);
            acc[nt] = __builtin_amdgcn_wmma_f32_16x16x32_f16(
                false, A, false, Bm, (short)0, acc[nt], false, false);
        }
    }
    #pragma unroll
    for (int nt = 0; nt < 4; ++nt) {
        int col = nt * 16 + row;
        float bc = bias[col];
        #pragma unroll
        for (int r = 0; r < 8; ++r) {
            int nrow = nbase + hi * 8 + r;
            float v = acc[nt][r] + bc;
            v = v > 0.f ? v : 0.f;
            v += agg[((size_t)b * NN + nrow) * ND + col];
            v = v > 0.f ? v : 0.f;
            h[((size_t)b * NN + nrow) * ND + col] = (_Float16)v;
        }
    }
}

// ---- scores = h @ final_W + final_b, masked ---------------------------------
__global__ void scores_kernel(const _Float16* __restrict__ h, const float* __restrict__ fW,
                              const float* __restrict__ fb, const float* __restrict__ x,
                              float* __restrict__ scores) {
    int idx = blockIdx.x * blockDim.x + threadIdx.x;
    if (idx >= NB * NN) return;
    float acc = fb[0];
    const _Float16* hp = h + (size_t)idx * ND;
    #pragma unroll 8
    for (int d = 0; d < ND; ++d) acc += (float)hp[d] * fW[d];
    bool sus = x[(size_t)idx * 3] > 0.5f;
    scores[idx] = sus ? -__builtin_inff() : acc;
}

// ---- one block per batch: lse[b] = max + log(sum exp(s - max)) --------------
__global__ void lse_kernel(const float* __restrict__ scores, float* __restrict__ lse) {
    __shared__ float red[256];
    int b = blockIdx.x;
    int tid = threadIdx.x;
    const float* s = scores + (size_t)b * NN;
    float m = -__builtin_inff();
    for (int i = tid; i < NN; i += 256) m = fmaxf(m, s[i]);
    red[tid] = m; __syncthreads();
    for (int w = 128; w > 0; w >>= 1) { if (tid < w) red[tid] = fmaxf(red[tid], red[tid + w]); __syncthreads(); }
    float gmax = red[0]; __syncthreads();
    float sum = 0.f;
    for (int i = tid; i < NN; i += 256) {
        float sc = s[i];
        if (sc > -__builtin_inff()) sum += expf(sc - gmax);
    }
    red[tid] = sum; __syncthreads();
    for (int w = 128; w > 0; w >>= 1) { if (tid < w) red[tid] += red[tid + w]; __syncthreads(); }
    if (tid == 0) lse[b] = gmax + logf(red[0]);
}

__global__ void out_kernel(const float* __restrict__ scores, const float* __restrict__ lse,
                           float* __restrict__ out) {
    int idx = blockIdx.x * blockDim.x + threadIdx.x;
    if (idx >= NB * NN) return;
    out[idx] = scores[idx] - lse[idx / NN];
}

extern "C" void kernel_launch(void* const* d_in, const int* in_sizes, int n_in,
                              void* d_out, int out_size, void* d_ws, size_t ws_size,
                              hipStream_t stream) {
    const float* x         = (const float*)d_in[0];
    const int*   edge_idx  = (const int*)d_in[1];
    const float* edge_attr = (const float*)d_in[2];
    const float* pv_W      = (const float*)d_in[3];
    const float* pv_b      = (const float*)d_in[4];
    const float* pe_W      = (const float*)d_in[5];
    const float* pe_b      = (const float*)d_in[6];
    const float* fe_W      = (const float*)d_in[7];
    const float* fe_b      = (const float*)d_in[8];
    const float* fv_W      = (const float*)d_in[9];
    const float* fv_b      = (const float*)d_in[10];
    const float* final_W   = (const float*)d_in[11];
    const float* final_b   = (const float*)d_in[12];
    float* out = (float*)d_out;

    const int* src = edge_idx;
    const int* dst = edge_idx + NE;

    // workspace carve-up (256B aligned)
    char* ws = (char*)d_ws;
    size_t off = 0;
    auto carve = [&](size_t bytes) { void* p = ws + off; off = (off + bytes + 255) & ~(size_t)255; return p; };
    _Float16* g      = (_Float16*)carve((size_t)NB * NE * ND * 2);   // 64 MB
    _Float16* h      = (_Float16*)carve((size_t)NB * NN * ND * 2);   // 12.8 MB
    float*    agg    = (float*)   carve((size_t)NB * NN * ND * 4);   // 25.6 MB
    _Float16* feWt   = (_Float16*)carve((size_t)NL * ND * 2 * ND * 2);
    _Float16* fvWt   = (_Float16*)carve((size_t)NL * ND * ND * 2);
    float*    scores = (float*)   carve((size_t)NB * NN * 4);
    float*    lse    = (float*)   carve(2 * sizeof(float));

    const size_t aggBytes = (size_t)NB * NN * ND * 4;

    // weight prep
    wconvert_kernel<<<(NL * ND * 2 * ND + 255) / 256, 256, 0, stream>>>(fe_W, fv_W, feWt, fvWt);

    // h0 / g0
    {
        long long nh = (long long)NB * NN * ND;
        init_h_kernel<<<(unsigned)((nh + 255) / 256), 256, 0, stream>>>(x, pv_W, pv_b, h);
        long long ng = (long long)NE * ND;
        init_g_kernel<<<(unsigned)((ng + 255) / 256), 256, 0, stream>>>(edge_attr, pe_W, pe_b, g);
    }

    // layers
    const int edgeWaves = NB * (NE / 16);
    const int edgeBlocks = (edgeWaves * 32 + 255) / 256;
    const int nodeWaves = NB * (NN / 16);
    const int nodeBlocks = (nodeWaves * 32 + 255) / 256;
    for (int l = 0; l < NL; ++l) {
        hipMemsetAsync(agg, 0, aggBytes, stream);
        edge_mlp_kernel<<<edgeBlocks, 256, 0, stream>>>(
            g, h, src, dst, feWt + (size_t)l * ND * 2 * ND, fe_b + (size_t)l * ND, agg);
        node_mlp_kernel<<<nodeBlocks, 256, 0, stream>>>(
            h, fvWt + (size_t)l * ND * ND, fv_b + (size_t)l * ND, agg);
    }

    // head + masked log-softmax
    scores_kernel<<<(NB * NN + 255) / 256, 256, 0, stream>>>(h, final_W, final_b, x, scores);
    lse_kernel<<<NB, 256, 0, stream>>>(scores, lse);
    out_kernel<<<(NB * NN + 255) / 256, 256, 0, stream>>>(scores, lse, out);
}
